// Recurrence_61160334295113
// MI455X (gfx1250) — matrix operand order repaired and checked by reading, not compiled, not run on hardware
//
#include <hip/hip_runtime.h>
#include <hip/hip_bf16.h>
#include <math.h>

#define T_   8
#define N_   64
#define HW_  256
#define HID_ 128
#define NA_  128
#define TL_  16
#define HX_  514
#define R_   512  // T*N

typedef __attribute__((ext_vector_type(16))) __bf16 v16bf;
typedef __attribute__((ext_vector_type(8)))  float  v8f;
typedef __attribute__((ext_vector_type(4)))  float  f32x4;
typedef __attribute__((ext_vector_type(4)))  unsigned int u32x4;

struct B32 { u32x4 lo, hi; };

// A-fragment (16x32 bf16, MxK): lane m=lane&15, half=lane>>4.
// elements 0..7  : K = kbase + half*8 + e        (contiguous 16B)
// elements 8..15 : K = kbase + 16 + half*8 + e-8 (contiguous 16B)
__device__ inline v16bf frag_a_rowmajor(const __bf16* rowptr, int kbase, int half) {
  B32 t;
  t.lo = *(const u32x4*)(rowptr + kbase + half * 8);
  t.hi = *(const u32x4*)(rowptr + kbase + 16 + half * 8);
  return __builtin_bit_cast(v16bf, t);
}

// B-fragment (32x16 bf16, KxN): lane col n=lane&15, half=lane>>4.
// elements 0..7 : K = kbase + half*8 + e; elements 8..15 : K = kbase + 16 + half*8 + e-8 — data stored K-contiguous per column.
__device__ inline v16bf frag_b_kcontig(const __bf16* colptr, int kbase, int half) {
  B32 t;
  const __bf16* p = colptr + kbase + half * 8;
  t.lo = *(const u32x4*)(p);
  t.hi = *(const u32x4*)(p + 16);
  return __builtin_bit_cast(v16bf, t);
}

__device__ inline v8f wmma_bf16(v16bf a, v16bf b, v8f c) {
  return __builtin_amdgcn_wmma_f32_16x16x32_bf16(false, a, false, b, (short)0, c, false, false);
}

// ---------------- gi = S @ w_ih^T + b_ih  (256 x 384) ----------------
__global__ void gi_kernel(const float* __restrict__ S, const float* __restrict__ w_ih,
                          const float* __restrict__ b_ih, float* __restrict__ gi) {
  __shared__ alignas(16) float Srow[HID_];
  int s = blockIdx.x, j = threadIdx.x;
  if (j < HID_) Srow[j] = S[s * HID_ + j];
  __syncthreads();
  const f32x4* wr = (const f32x4*)(w_ih + j * HID_);
  const f32x4* hv = (const f32x4*)Srow;
  float acc = b_ih[j];
#pragma unroll 8
  for (int k = 0; k < HID_ / 4; ++k) {
    f32x4 w4 = wr[k], h4 = hv[k];
    acc += w4.x * h4.x + w4.y * h4.y + w4.z * h4.z + w4.w * h4.w;
  }
  gi[s * 384 + j] = acc;
}

// ---------------- GRU scan (serial over 256 steps), emit K in bf16 ----------------
__global__ void gru_kernel(const float* __restrict__ gi, const float* __restrict__ w_hh,
                           const float* __restrict__ b_hh, __bf16* __restrict__ Kb) {
  __shared__ alignas(16) float hsh[HID_];
  __shared__ float gsh[3 * HID_];
  int j = threadIdx.x;
  if (j < HID_) hsh[j] = 0.f;
  __syncthreads();
  const f32x4* wr = (const f32x4*)(w_hh + j * HID_);
  float bj = b_hh[j];
  for (int s = 0; s < HW_; ++s) {
    const f32x4* hv = (const f32x4*)hsh;
    float acc = bj;
#pragma unroll 8
    for (int k = 0; k < HID_ / 4; ++k) {
      f32x4 w4 = wr[k], h4 = hv[k];
      acc += w4.x * h4.x + w4.y * h4.y + w4.z * h4.z + w4.w * h4.w;
    }
    gsh[j] = acc;
    __syncthreads();
    if (j < HID_) {
      const float* git = gi + s * 384;
      float r = 1.f / (1.f + expf(-(git[j] + gsh[j])));
      float z = 1.f / (1.f + expf(-(git[HID_ + j] + gsh[HID_ + j])));
      float n = tanhf(git[2 * HID_ + j] + r * gsh[2 * HID_ + j]);
      float hn = (1.f - z) * n + z * hsh[j];
      hsh[j] = hn;
      Kb[s * HID_ + j] = (__bf16)hn;
    }
    __syncthreads();
  }
}

// ---------------- f32 -> bf16 conversion ----------------
__global__ void cvt_bf16_kernel(const float* __restrict__ src, __bf16* __restrict__ dst, int n) {
  int i = blockIdx.x * blockDim.x + threadIdx.x;
  if (i < n) dst[i] = (__bf16)src[i];
}

// ---------------- MLP: y2 = relu(relu((A[a]+S[s])@W1^T+b1)@W2^T+b2) ----------------
// Output stored as Y2t[w=s][a0][a2] bf16 (a2 contiguous) for logits B-fragments.
__global__ void mlp_kernel(const float* __restrict__ Amat, const float* __restrict__ S,
                           const __bf16* __restrict__ W1b, const float* __restrict__ b1,
                           const __bf16* __restrict__ W2b, const float* __restrict__ b2,
                           __bf16* __restrict__ Y2t) {
  __shared__ alignas(16) __bf16 y1s[4][16][HID_];
  int tid = threadIdx.x;
  int wave = tid >> 5, lane = tid & 31, m = lane & 15, half = lane >> 4;
  int RT = blockIdx.x * 4 + wave;   // 2048 row-tiles of 16 rows; row = a0*256 + s
  int a0 = RT >> 4;
  int s0 = (RT & 15) * 16;
  const float* Ar = Amat + a0 * HID_;
  const float* Sr = S + (s0 + m) * HID_;

  v8f acc[8];
#pragma unroll
  for (int ct = 0; ct < 8; ++ct) acc[ct] = (v8f){};
  // GEMM1: X = A[a0] + S[s0+m] built directly as A-fragments
#pragma unroll
  for (int ks = 0; ks < 4; ++ks) {
    int kbase = ks * 32;
    float xa[16], xs[16];
    const f32x4* Ap = (const f32x4*)(Ar + kbase);
    const f32x4* Sp = (const f32x4*)(Sr + kbase);
    ((f32x4*)xa)[0] = Ap[half * 2];     ((f32x4*)xa)[1] = Ap[half * 2 + 1];
    ((f32x4*)xa)[2] = Ap[4 + half * 2]; ((f32x4*)xa)[3] = Ap[5 + half * 2];
    ((f32x4*)xs)[0] = Sp[half * 2];     ((f32x4*)xs)[1] = Sp[half * 2 + 1];
    ((f32x4*)xs)[2] = Sp[4 + half * 2]; ((f32x4*)xs)[3] = Sp[5 + half * 2];
    v16bf afrag;
#pragma unroll
    for (int e = 0; e < 16; ++e) afrag[e] = (__bf16)(xa[e] + xs[e]);
#pragma unroll
    for (int ct = 0; ct < 8; ++ct) {
      v16bf bfrag = frag_b_kcontig(W1b + (ct * 16 + m) * HID_, kbase, half);
      acc[ct] = wmma_bf16(afrag, bfrag, acc[ct]);
    }
  }
  // bias + relu -> LDS (re-layout C-fragment -> A-fragment source)
#pragma unroll
  for (int ct = 0; ct < 8; ++ct) {
    float bb = b1[ct * 16 + m];
#pragma unroll
    for (int i = 0; i < 8; ++i) {
      float v = acc[ct][i] + bb;
      y1s[wave][i + 8 * half][ct * 16 + m] = (__bf16)fmaxf(v, 0.f);
    }
  }
  __syncthreads();
  // GEMM2
  v8f acc2[8];
#pragma unroll
  for (int ct = 0; ct < 8; ++ct) acc2[ct] = (v8f){};
#pragma unroll
  for (int ks = 0; ks < 4; ++ks) {
    int kbase = ks * 32;
    v16bf afrag = frag_a_rowmajor(&y1s[wave][m][0], kbase, half);
#pragma unroll
    for (int ct = 0; ct < 8; ++ct) {
      v16bf bfrag = frag_b_kcontig(W2b + (ct * 16 + m) * HID_, kbase, half);
      acc2[ct] = wmma_bf16(afrag, bfrag, acc2[ct]);
    }
  }
  // bias + relu -> Y2t[s][a0][a2]
#pragma unroll
  for (int ct = 0; ct < 8; ++ct) {
    float bb = b2[ct * 16 + m];
#pragma unroll
    for (int i = 0; i < 8; ++i) {
      float v = fmaxf(acc2[ct][i] + bb, 0.f);
      int s = s0 + i + 8 * half;
      Y2t[(size_t)s * (NA_ * HID_) + a0 * NA_ + (ct * 16 + m)] = (__bf16)v;
    }
  }
}

// ---------------- logits_w = K @ B_w, softmax over s, store Pt[w][a][s] bf16 ----------------
__global__ void logits_softmax_kernel(const __bf16* __restrict__ Kb,
                                      const __bf16* __restrict__ Y2t,
                                      __bf16* __restrict__ Pt) {
  extern __shared__ char smem[];
  float* lg = (float*)smem;  // [256][128]
  int w = blockIdx.x;
  int tid = threadIdx.x;
  int wave = tid >> 5, lane = tid & 31, m = lane & 15, half = lane >> 4;
  const __bf16* Bw = Y2t + (size_t)w * (NA_ * HID_);
#pragma unroll
  for (int sub = 0; sub < 2; ++sub) {
    int s0 = (wave * 2 + sub) * 16;
    v8f acc[8];
#pragma unroll
    for (int ct = 0; ct < 8; ++ct) acc[ct] = (v8f){};
#pragma unroll
    for (int ks = 0; ks < 4; ++ks) {
      int kbase = ks * 32;
      v16bf afrag = frag_a_rowmajor(Kb + (s0 + m) * HID_, kbase, half);
#pragma unroll
      for (int ct = 0; ct < 8; ++ct) {
        v16bf bfrag = frag_b_kcontig(Bw + (ct * 16 + m) * HID_, kbase, half);
        acc[ct] = wmma_bf16(afrag, bfrag, acc[ct]);
      }
    }
#pragma unroll
    for (int ct = 0; ct < 8; ++ct)
#pragma unroll
      for (int i = 0; i < 8; ++i)
        lg[(s0 + i + 8 * half) * NA_ + ct * 16 + m] = acc[ct][i];
  }
  __syncthreads();
  if (tid < NA_) {
    int a = tid;
    float mx = -3.4e38f;
    for (int s = 0; s < HW_; ++s) mx = fmaxf(mx, lg[s * NA_ + a]);
    float sum = 0.f;
    for (int s = 0; s < HW_; ++s) sum += expf(lg[s * NA_ + a] - mx);
    float inv = 1.f / sum;
    __bf16* outp = Pt + (size_t)w * (NA_ * HW_) + (size_t)a * HW_;
    for (int s = 0; s < HW_; ++s) outp[s] = (__bf16)(expf(lg[s * NA_ + a] - mx) * inv);
  }
}

// ---------------- Bellman iteration 1 degenerates to values = rewards ----------------
__global__ void initv_kernel(const float* __restrict__ inputs, __bf16* __restrict__ Vb) {
  int r = blockIdx.x, w = threadIdx.x;
  Vb[r * HW_ + w] = (__bf16)inputs[r * (HW_ + 1) + w];
}

// ---------------- One Bellman iteration: V' = R + max_a (V @ P_w) ----------------
__global__ void bellman_kernel(const __bf16* __restrict__ Vin,
                               const __bf16* __restrict__ Pt,
                               const float* __restrict__ inputs,
                               __bf16* __restrict__ Vout,
                               float* __restrict__ Vf) {
  extern __shared__ char smem[];
  __bf16* Pl = (__bf16*)smem;  // Pt_w staged: [NA_][HW_] bf16 = 64 KB
  int w = blockIdx.x, tid = threadIdx.x;
  {
    const u32x4* src = (const u32x4*)(Pt + (size_t)w * (NA_ * HW_));
    u32x4* dst = (u32x4*)smem;
    for (int i = tid; i < (NA_ * HW_ * 2) / 16; i += blockDim.x) dst[i] = src[i];
  }
  __syncthreads();
  int wave = tid >> 5, lane = tid & 31, m = lane & 15, half = lane >> 4;
#pragma unroll 1
  for (int rt8 = 0; rt8 < 4; ++rt8) {
    int rbase = (wave * 4 + rt8) * 16;
    v8f acc[8];
#pragma unroll
    for (int ct = 0; ct < 8; ++ct) acc[ct] = (v8f){};
    const __bf16* arow = Vin + (rbase + m) * HW_;
#pragma unroll
    for (int ks = 0; ks < 8; ++ks) {
      int kbase = ks * 32;
      v16bf afrag = frag_a_rowmajor(arow, kbase, half);
#pragma unroll
      for (int ct = 0; ct < 8; ++ct) {
        v16bf bfrag = frag_b_kcontig(Pl + (ct * 16 + m) * HW_, kbase, half);
        acc[ct] = wmma_bf16(afrag, bfrag, acc[ct]);
      }
    }
    // row-max over a (8 col-tiles in regs, then 16 lanes of same half)
#pragma unroll
    for (int i = 0; i < 8; ++i) {
      float mx = acc[0][i];
#pragma unroll
      for (int ct = 1; ct < 8; ++ct) mx = fmaxf(mx, acc[ct][i]);
      for (int d = 1; d < 16; d <<= 1) mx = fmaxf(mx, __shfl_xor(mx, d, 16));
      if (m == 0) {
        int row = rbase + i + 8 * half;
        float v = inputs[row * (HW_ + 1) + w] + mx;
        Vf[row * HW_ + w] = v;
        Vout[row * HW_ + w] = (__bf16)v;
      }
    }
  }
}

// ---------------- Assemble output: concat(passthrough, values), plus out[-1:] ----------------
__global__ void out_kernel(const float* __restrict__ rnn_hxs,
                           const float* __restrict__ Vf,
                           float* __restrict__ out) {
  int idx = blockIdx.x * blockDim.x + threadIdx.x;
  if (idx >= T_ * N_ * HX_) return;
  int c = idx % HX_;
  int tn = idx / HX_;
  float v = (c < 258) ? rnn_hxs[idx] : Vf[tn * HW_ + (c - 258)];
  out[idx] = v;
  int t = tn / N_;
  if (t == T_ - 1) out[T_ * N_ * HX_ + (tn - (T_ - 1) * N_) * HX_ + c] = v;
}

extern "C" void kernel_launch(void* const* d_in, const int* in_sizes, int n_in,
                              void* d_out, int out_size, void* d_ws, size_t ws_size,
                              hipStream_t stream) {
  (void)in_sizes; (void)n_in; (void)out_size; (void)ws_size;
  const float* inputs  = (const float*)d_in[0];
  const float* rnn_hxs = (const float*)d_in[1];
  const float* S       = (const float*)d_in[2];
  const float* A       = (const float*)d_in[3];
  const float* w_ih    = (const float*)d_in[4];
  const float* w_hh    = (const float*)d_in[5];
  const float* b_ih    = (const float*)d_in[6];
  const float* b_hh    = (const float*)d_in[7];
  const float* W1      = (const float*)d_in[8];
  const float* b1      = (const float*)d_in[9];
  const float* W2      = (const float*)d_in[10];
  const float* b2      = (const float*)d_in[11];
  float* out = (float*)d_out;

  char* ws = (char*)d_ws;
  size_t o = 0;
  auto take = [&](size_t bytes) {
    char* p = ws + o;
    o = (o + bytes + 255) & ~(size_t)255;
    return p;
  };
  float*  gi  = (float*)take((size_t)HW_ * 384 * 4);
  __bf16* Kb  = (__bf16*)take((size_t)HW_ * HID_ * 2);
  __bf16* W1b = (__bf16*)take((size_t)HID_ * HID_ * 2);
  __bf16* W2b = (__bf16*)take((size_t)NA_ * HID_ * 2);
  __bf16* Y2t = (__bf16*)take((size_t)HW_ * NA_ * HID_ * 2);
  __bf16* Pt  = (__bf16*)take((size_t)HW_ * NA_ * HW_ * 2);
  __bf16* VbA = (__bf16*)take((size_t)R_ * HW_ * 2);
  __bf16* VbB = (__bf16*)take((size_t)R_ * HW_ * 2);
  float*  Vf  = (float*)take((size_t)R_ * HW_ * 4);

  gi_kernel<<<HW_, 384, 0, stream>>>(S, w_ih, b_ih, gi);
  gru_kernel<<<1, 384, 0, stream>>>(gi, w_hh, b_hh, Kb);
  cvt_bf16_kernel<<<(HID_ * HID_ + 255) / 256, 256, 0, stream>>>(W1, W1b, HID_ * HID_);
  cvt_bf16_kernel<<<(NA_ * HID_ + 255) / 256, 256, 0, stream>>>(W2, W2b, NA_ * HID_);
  mlp_kernel<<<512, 128, 0, stream>>>(A, S, W1b, b1, W2b, b2, Y2t);
  logits_softmax_kernel<<<HW_, 256, HW_ * NA_ * sizeof(float), stream>>>(Kb, Y2t, Pt);
  initv_kernel<<<R_, HW_, 0, stream>>>(inputs, VbA);
  __bf16* vin = VbA;
  __bf16* vout = VbB;
  for (int it = 0; it < TL_ - 1; ++it) {
    bellman_kernel<<<HW_, 256, NA_ * HW_ * 2, stream>>>(vin, Pt, inputs, vout, Vf);
    __bf16* t = vin; vin = vout; vout = t;
  }
  out_kernel<<<(T_ * N_ * HX_ + 255) / 256, 256, 0, stream>>>(rnn_hxs, Vf, out);
}
